// NCA_81999515615286
// MI455X (gfx1250) — matrix operand order repaired and checked by reading, not compiled, not run on hardware
//
#include <hip/hip_runtime.h>

// ---------------------------------------------------------------------------
// NCA update step for MI455X (gfx1250, wave32, WMMA).
//
// Roofline: 17.4 GFLOP MLP vs ~120 MB mandatory HBM traffic (~6us @23.3TB/s).
// fp16 WMMA (16x16x32, f32 accum) makes the GEMMs free relative to memory,
// so everything (perception + GEMM1 + GEMM2 + stochastic residual) is fused
// into one LDS-staged kernel; a second tiny kernel applies the alive mask
// (needs maxpool of the *updated* state => cross-workgroup dependency).
// ---------------------------------------------------------------------------

typedef __attribute__((ext_vector_type(16))) _Float16 v16h;
typedef __attribute__((ext_vector_type(8)))  _Float16 v8h;
typedef __attribute__((ext_vector_type(8)))  float    v8f;

#define ALIVE_THRESH 0.1f
#define STEP         0.1f
#define UPD_PROB     0.5f

// B=16, H=W=256, C=14, CIN=17, K1=51 (pad 64), HID=128, tile = 128 pixels/WG

__global__ __launch_bounds__(256)
void nca_update_kernel(const float* __restrict__ x, const float* __restrict__ img,
                       const float* __restrict__ rnd, const float* __restrict__ w1,
                       const float* __restrict__ b1, const float* __restrict__ w2,
                       const float* __restrict__ b2, float* __restrict__ out,
                       float* __restrict__ ch0ws, float* __restrict__ alivews)
{
    // LDS: 13260 + 16384 + 32768 = 62412 bytes
    __shared__ _Float16 xc[3 * 130 * 17];                 // halo'd input slab (fp16)
    __shared__ __align__(16) _Float16 percep[128 * 64];   // perception, K padded 51->64
    __shared__ __align__(16) _Float16 hbuf[128 * 128];    // hidden activations

    const int tid   = threadIdx.x;
    const int lane  = tid & 31;
    const int wave  = tid >> 5;          // 8 waves
    const int ln    = lane & 15;         // N index within 16-wide tile
    const int lhalf = lane >> 4;         // wave half (WMMA operand layout)

    const int blk  = blockIdx.x;         // 16 * 256 * 2 workgroups
    const int half = blk & 1;
    const int h    = (blk >> 1) & 255;
    const int b    = blk >> 9;
    const int c0   = half * 128;

    // ---- Phase 1: cooperative load of (x ++ image) rows h-1..h+1 with halo ----
    for (int idx = tid; idx < 3 * 130 * 17; idx += 256) {
        int c    = idx % 17;
        int rest = idx / 17;
        int col  = rest % 130;
        int r    = rest / 130;
        int gr   = h + r - 1;
        int gc   = c0 + col - 1;
        float v  = 0.f;                                  // zero pad ('SAME' conv)
        if ((unsigned)gr < 256u && (unsigned)gc < 256u) {
            v = (c < 14) ? x[(((b * 256 + gr) * 256) + gc) * 14 + c]
                         : img[((gr * 256) + gc) * 3 + (c - 14)];
        }
        xc[idx] = (_Float16)v;
    }
    __syncthreads();

    // ---- Phase 2: perception (identity + Sobel dx/dy), alive_before ----
    for (int idx = tid; idx < 128 * 17; idx += 256) {
        int c  = idx % 17;
        int p  = idx / 17;
        int sp = p + 1;
        float v00 = (float)xc[((0*130)+sp-1)*17+c], v01 = (float)xc[((0*130)+sp)*17+c], v02 = (float)xc[((0*130)+sp+1)*17+c];
        float v10 = (float)xc[((1*130)+sp-1)*17+c], v11 = (float)xc[((1*130)+sp)*17+c], v12 = (float)xc[((1*130)+sp+1)*17+c];
        float v20 = (float)xc[((2*130)+sp-1)*17+c], v21 = (float)xc[((2*130)+sp)*17+c], v22 = (float)xc[((2*130)+sp+1)*17+c];
        float ident = v11;
        float dx = ((v02 - v00) + 2.f * (v12 - v10) + (v22 - v20)) * 0.125f;
        float dy = ((v20 + 2.f * v21 + v22) - (v00 + 2.f * v01 + v02)) * 0.125f;
        int base = p * 64 + c * 3;                       // TF depthwise order: o = c*3 + k
        percep[base + 0] = (_Float16)ident;
        percep[base + 1] = (_Float16)dx;
        percep[base + 2] = (_Float16)dy;
    }
    for (int idx = tid; idx < 128 * 13; idx += 256)      // zero K padding 51..63
        percep[(idx / 13) * 64 + 51 + (idx % 13)] = (_Float16)0.f;
    if (tid < 128) {                                     // alive_before = maxpool3(x ch0) > T
        int sp = tid + 1;
        float m = -1e30f;
        #pragma unroll
        for (int r = 0; r < 3; ++r)
            #pragma unroll
            for (int j = -1; j <= 1; ++j)
                m = fmaxf(m, (float)xc[((r * 130) + sp + j) * 17 + 0]);
        int pflat = ((b * 256 + h) * 256) + c0 + tid;
        alivews[pflat] = (m > ALIVE_THRESH) ? 1.f : 0.f;
    }
    __syncthreads();

    // ---- Phase 3: GEMM1  h = relu(percep @ w1 + b1); wave owns 16 cols of 128 ----
    union Frag { v16h v; _Float16 e[16]; };
    const int n0 = wave * 16;
    Frag bw1[2];                                         // w1 B-fragments resident in VGPRs
    #pragma unroll
    for (int ks = 0; ks < 2; ++ks)
        #pragma unroll
        for (int e = 0; e < 16; ++e) {
            int k = ks * 32 + lhalf * 8 + (e < 8 ? e : e + 8); // B layout: lanes0-15 K=0..7 / 16..23, 16-31 K=8..15 / 24..31
            bw1[ks].e[e] = (_Float16)((k < 51) ? w1[k * 128 + n0 + ln] : 0.f);
        }
    float b1c = b1[n0 + ln];

    for (int mt = 0; mt < 8; ++mt) {
        int pix = mt * 16 + ln;
        v8f acc = {};
        #pragma unroll
        for (int ks = 0; ks < 2; ++ks) {
            // A layout: lanes0-15 K=0..7 / 16..23, lanes16-31 K=8..15 / 24..31
            const v8h* a0 = (const v8h*)&percep[pix * 64 + ks * 32 + lhalf * 8];
            const v8h* a1 = (const v8h*)&percep[pix * 64 + ks * 32 + lhalf * 8 + 16];
            v16h a = __builtin_shufflevector(*a0, *a1, 0,1,2,3,4,5,6,7,8,9,10,11,12,13,14,15);
            acc = __builtin_amdgcn_wmma_f32_16x16x32_f16(false, a, false, bw1[ks].v,
                                                         (short)0, acc, false, false);
        }
        #pragma unroll
        for (int r = 0; r < 8; ++r) {                    // C layout: VGPR r -> M=r (+8 for hi half)
            int m = r + lhalf * 8;
            float hv = fmaxf(acc[r] + b1c, 0.f);
            hbuf[(mt * 16 + m) * 128 + n0 + ln] = (_Float16)hv;
        }
    }
    __syncthreads();

    // ---- Phase 4: GEMM2  dxp = (h @ w2 + b2)*STEP; stochastic residual -> xn ----
    Frag bw2[4];
    #pragma unroll
    for (int ks = 0; ks < 4; ++ks)
        #pragma unroll
        for (int e = 0; e < 16; ++e) {
            int k = ks * 32 + lhalf * 8 + (e < 8 ? e : e + 8);
            bw2[ks].e[e] = (_Float16)((ln < 14) ? w2[k * 14 + ln] : 0.f);  // N padded 14->16
        }
    float b2c = (ln < 14) ? b2[ln] : 0.f;

    {
        int mt  = wave;                                  // 8 M-tiles, one per wave
        int pix = mt * 16 + ln;
        v8f acc = {};
        #pragma unroll
        for (int ks = 0; ks < 4; ++ks) {
            const v8h* a0 = (const v8h*)&hbuf[pix * 128 + ks * 32 + lhalf * 8];
            const v8h* a1 = (const v8h*)&hbuf[pix * 128 + ks * 32 + lhalf * 8 + 16];
            v16h a = __builtin_shufflevector(*a0, *a1, 0,1,2,3,4,5,6,7,8,9,10,11,12,13,14,15);
            acc = __builtin_amdgcn_wmma_f32_16x16x32_f16(false, a, false, bw2[ks].v,
                                                         (short)0, acc, false, false);
        }
        #pragma unroll
        for (int r = 0; r < 8; ++r) {
            int m     = r + lhalf * 8;
            int pflat = ((b * 256 + h) * 256) + c0 + mt * 16 + m;
            float dxp = (acc[r] + b2c) * STEP;
            float u   = (rnd[pflat] <= UPD_PROB) ? 1.f : 0.f;
            if (ln < 14) {
                float xn = x[pflat * 14 + ln] + dxp * u; // lanes 0..13: contiguous 56B store
                out[pflat * 14 + ln] = xn;
                if (ln == 0) ch0ws[pflat] = xn;
            }
        }
    }
}

// ---- Kernel B: alive = alive_before && maxpool3(xn ch0) > T; zero dead pixels ----
__global__ __launch_bounds__(256)
void nca_mask_kernel(float* __restrict__ out, const float* __restrict__ ch0,
                     const float* __restrict__ alivebef)
{
    int p  = blockIdx.x * 256 + threadIdx.x;             // 0 .. 16*256*256-1
    int b  = p >> 16;
    int hh = (p >> 8) & 255;
    int wc = p & 255;
    float m = -1e30f;
    #pragma unroll
    for (int dh = -1; dh <= 1; ++dh) {
        int nh = hh + dh;
        if ((unsigned)nh >= 256u) continue;              // 'SAME' pool: ignore outside
        #pragma unroll
        for (int dw = -1; dw <= 1; ++dw) {
            int nw = wc + dw;
            if ((unsigned)nw >= 256u) continue;
            m = fmaxf(m, ch0[(b << 16) + nh * 256 + nw]);
        }
    }
    bool alive = (alivebef[p] > 0.5f) && (m > ALIVE_THRESH);
    if (!alive) {                                        // write only when dead
        #pragma unroll
        for (int c = 0; c < 14; ++c) out[p * 14 + c] = 0.f;
    }
}

extern "C" void kernel_launch(void* const* d_in, const int* in_sizes, int n_in,
                              void* d_out, int out_size, void* d_ws, size_t ws_size,
                              hipStream_t stream) {
    const float* x   = (const float*)d_in[0];
    const float* img = (const float*)d_in[1];
    const float* rnd = (const float*)d_in[2];
    const float* w1  = (const float*)d_in[3];
    const float* b1  = (const float*)d_in[4];
    const float* w2  = (const float*)d_in[5];
    const float* b2  = (const float*)d_in[6];
    float* out = (float*)d_out;
    float* ch0 = (float*)d_ws;                           // xn channel-0 plane (4 MB)
    float* alv = ch0 + 16 * 256 * 256;                   // alive_before plane (4 MB)

    nca_update_kernel<<<16 * 256 * 2, 256, 0, stream>>>(x, img, rnd, w1, b1, w2, b2,
                                                        out, ch0, alv);
    nca_mask_kernel<<<(16 * 256 * 256) / 256, 256, 0, stream>>>(out, ch0, alv);
}
